// CorefClusterer_72825465471240
// MI455X (gfx1250) — matrix operand order repaired and checked by reading, not compiled, not run on hardware
//
#include <hip/hip_runtime.h>
#include <hip/hip_bf16.h>
#include <math.h>
#include <stdint.h>

// ---------------- problem constants (fixed by reference setup_inputs) -------
#define NW    2048            // n words
#define DW    512             // d
#define G4D   (4 * DW)        // 2048 gate width
#define KTOP  50
#define DEMB  64
#define PAIRD (3 * DW + DEMB) // 1600
#define HID   512
#define FEAT_LD 1608          // padded LDS row stride (16B-aligned, conflict-skewed)
#define TROWS 32              // fine-scorer rows per block (2 WMMA row-tiles)
#define FEATS_BYTES (TROWS * FEAT_LD * 2)   // 102912
#define BCHUNK_ELEMS (HID * 32)             // one 32-K hid_W chunk: 512x32 bf16 = 32KB
#define EPSILONV 1e-07f
#define NEG_SLOPE 0.01f

typedef __attribute__((ext_vector_type(16))) __bf16        v16bf;
typedef __attribute__((ext_vector_type(8)))  float         v8f;
typedef __attribute__((ext_vector_type(4)))  unsigned int  v4u;
typedef __attribute__((ext_vector_type(4)))  unsigned int  u32x4;
typedef __attribute__((ext_vector_type(8)))  int           i32x8;
typedef __attribute__((ext_vector_type(4)))  int           i32x4;

#if defined(__has_builtin)
#if __has_builtin(__builtin_amdgcn_tensor_load_to_lds) && __has_builtin(__builtin_amdgcn_s_wait_tensorcnt)
#define HAVE_TDM 1
#endif
#endif
#ifndef HAVE_TDM
#define HAVE_TDM 0
#endif

// ---------------- helpers ---------------------------------------------------
static __device__ __forceinline__ unsigned short f2bf(float f) {
  union { float f; unsigned int u; } v; v.f = f;
  unsigned int r = v.u + 0x7fffu + ((v.u >> 16) & 1u);  // round-to-nearest-even
  return (unsigned short)(r >> 16);
}

// A-operand fragment (16x32 bf16, ISA 7.12.2: lanes 0-15 K{0..7,16..23},
// lanes 16-31 K{8..15,24..31}); two b128 loads per lane.
static __device__ __forceinline__ v16bf
load_frag_a(const unsigned short* base, int ld, int row0, int k0, int lane) {
  int m    = row0 + (lane & 15);
  int half = (lane >> 4) & 1;
  const unsigned short* p = base + (size_t)m * ld + k0 + half * 8;
  union { v16bf v; v4u q[2]; } u;
  u.q[0] = *(const v4u*)(p);
  u.q[1] = *(const v4u*)(p + 16);
  return u.v;
}

// B-operand fragment (32x16 bf16; A*B^T with B row-major [N,K]: same K order
// as A, lanes 0-15 K{0..7,16..23}, lanes 16-31 K{8..15,24..31}).
static __device__ __forceinline__ v16bf
load_frag_b(const unsigned short* base, int ld, int col0, int k0, int lane) {
  int nn   = col0 + (lane & 15);
  int half = (lane >> 4) & 1;
  const unsigned short* p = base + (size_t)nn * ld + k0 + half * 8;
  union { v16bf v; v4u q[2]; } u;
  u.q[0] = *(const v4u*)(p);
  u.q[1] = *(const v4u*)(p + 16);
  return u.v;
}

static __device__ __forceinline__ v8f
wmma_bf16(v16bf a, v16bf b, v8f c) {
  return __builtin_amdgcn_wmma_f32_16x16x32_bf16(false, a, false, b, (short)0, c, false, false);
}

static __device__ __forceinline__ float sigmoidf_(float x) {
  return 1.0f / (1.0f + __expf(-x));
}

#if HAVE_TDM
// Issue a TDM DMA of one [512 rows x 32 K] bf16 chunk of hid_W into LDS.
// D# built per ISA ch.8: group0 = {count, lds_addr, global_addr, type=2},
// group1 = {data_size=2B, tensor dims, tile dims, dim0 stride}.
static __device__ __forceinline__ void
tdm_load_hidw_chunk(const unsigned short* hidwbf, int k0, unsigned short* ldsdst) {
  unsigned long long ga = (unsigned long long)(uintptr_t)hidwbf
                        + (unsigned long long)k0 * 2ull;
  unsigned int lds_off = (unsigned int)(uintptr_t)ldsdst;  // LDS aperture: low 32b = offset
  u32x4 g0;
  g0[0] = 1u;                                        // count=1, user descriptor
  g0[1] = lds_off;                                   // lds_addr [63:32]
  g0[2] = (unsigned int)(ga & 0xffffffffull);        // global_addr lo
  g0[3] = (unsigned int)((ga >> 32) & 0x01ffffffull) // global_addr hi [120:96]
        | (2u << 30);                                // type=2 ("image") [127:126]
  i32x8 g1;
  g1[0] = 1 << 16;                     // data_size=2B [17:16]; mask=0
  g1[1] = (PAIRD & 0xffff) << 16;      // tensor_dim0[15:0] @ bits 63:48 (=1600)
  g1[2] = (HID & 0xffff) << 16;        // tensor_dim1[15:0] @ bits 95:80 (=512)
  g1[3] = 32 << 16;                    // tile_dim0 @ bits 127:112 (=32 elems/row)
  g1[4] = HID;                         // tile_dim1 @ bits 143:128 (=512 rows)
  g1[5] = PAIRD;                       // tensor_dim0_stride[31:0] (=1600 elems)
  g1[6] = 0;
  g1[7] = 0;
  i32x4 gz = {0, 0, 0, 0};
#if __clang_major__ >= 23
  i32x8 gp = {0, 0, 0, 0, 0, 0, 0, 0};
  __builtin_amdgcn_tensor_load_to_lds(g0, g1, gz, gz, gp, 0);
#else
  __builtin_amdgcn_tensor_load_to_lds(g0, g1, gz, gz, 0);
#endif
}
#endif

// ---------------- tiny prep kernels ----------------------------------------
__global__ void f32_to_bf16_kernel(const float* __restrict__ src,
                                   unsigned short* __restrict__ dst, int count) {
  int i = blockIdx.x * blockDim.x + threadIdx.x;
  if (i < count) dst[i] = f2bf(src[i]);
}

__global__ void bias_sum_kernel(const float* __restrict__ a,
                                const float* __restrict__ b,
                                float* __restrict__ out, int count) {
  int i = blockIdx.x * blockDim.x + threadIdx.x;
  if (i < count) out[i] = a[i] + b[i];
}

// ---------------- generic WMMA GEMM: C[M,N] = A[M,K] * B[N,K]^T + bias ------
// 2x2 register-blocked: one wave computes a 32x32 C tile (4 wmma / 4 frag
// loads per K-step -> 2x the arithmetic intensity of 1x1 tiling).
static __device__ __forceinline__ void
store_tile(float* C, int N, int row0, int col0, int lane, v8f acc,
           const float* bias, int mask_mode) {
  const int nn    = col0 + (lane & 15);
  const int mbase = row0 + ((lane >> 4) << 3);
  const float bv  = bias ? bias[nn] : 0.0f;
#pragma unroll
  for (int r = 0; r < 8; ++r) {
    const int mm = mbase + r;
    float v = acc[r] + bv;
    if (mask_mode && !(mm > nn)) v = -__builtin_inff();
    C[(size_t)mm * N + nn] = v;
  }
}

__global__ void gemm_nt_bf16_wmma(const unsigned short* __restrict__ Abf, int lda,
                                  const unsigned short* __restrict__ Bbf, int ldb,
                                  const float* __restrict__ bias,
                                  float* __restrict__ C,
                                  int M, int N, int K, int mask_mode) {
  const int lane = threadIdx.x & 31;
  const int wave = threadIdx.x >> 5;
  const int wpb  = blockDim.x >> 5;
  const int row0 = (blockIdx.y * wpb + wave) * 32;
  const int col0 = blockIdx.x * 32;
  if (row0 >= M || col0 >= N) return;

  v8f acc00 = {}, acc01 = {}, acc10 = {}, acc11 = {};
  for (int k0 = 0; k0 < K; k0 += 32) {
    const v16bf a0 = load_frag_a(Abf, lda, row0,      k0, lane);
    const v16bf a1 = load_frag_a(Abf, lda, row0 + 16, k0, lane);
    const v16bf b0 = load_frag_b(Bbf, ldb, col0,      k0, lane);
    const v16bf b1 = load_frag_b(Bbf, ldb, col0 + 16, k0, lane);
    acc00 = wmma_bf16(a0, b0, acc00);
    acc01 = wmma_bf16(a0, b1, acc01);
    acc10 = wmma_bf16(a1, b0, acc10);
    acc11 = wmma_bf16(a1, b1, acc11);
  }
  store_tile(C, N, row0,      col0,      lane, acc00, bias, mask_mode);
  store_tile(C, N, row0,      col0 + 16, lane, acc01, bias, mask_mode);
  store_tile(C, N, row0 + 16, col0,      lane, acc10, bias, mask_mode);
  store_tile(C, N, row0 + 16, col0 + 16, lane, acc11, bias, mask_mode);
}

// ---------------- sequential LSTM recurrence (latency-bound; one WGP) -------
__global__ void lstm_recurrence_kernel(const float* __restrict__ xp,
                                       const float* __restrict__ W_hh,
                                       float* __restrict__ hs) {
  __shared__ float h[DW];
  __shared__ float c[DW];
  const int j = threadIdx.x;            // one hidden unit per thread
  h[j] = 0.f; c[j] = 0.f;
  __syncthreads();
  const float* wi = W_hh + (size_t)(0 * DW + j) * DW;
  const float* wf = W_hh + (size_t)(1 * DW + j) * DW;
  const float* wg = W_hh + (size_t)(2 * DW + j) * DW;
  const float* wo = W_hh + (size_t)(3 * DW + j) * DW;
  for (int t = 0; t < NW; ++t) {
    const float* xpt = xp + (size_t)t * G4D;
    if (t + 1 < NW) __builtin_prefetch(xp + (size_t)(t + 1) * G4D + j, 0, 1);
    float gi = xpt[j], gf = xpt[DW + j], gg = xpt[2 * DW + j], go = xpt[3 * DW + j];
#pragma unroll 8
    for (int kk = 0; kk < DW; ++kk) {
      const float hv = h[kk];
      gi = fmaf(wi[kk], hv, gi);
      gf = fmaf(wf[kk], hv, gf);
      gg = fmaf(wg[kk], hv, gg);
      go = fmaf(wo[kk], hv, go);
    }
    const float iv = sigmoidf_(gi), fv = sigmoidf_(gf), ov = sigmoidf_(go);
    const float gv = tanhf(gg);
    const float cn = fv * c[j] + iv * gv;
    const float hn = ov * tanhf(cn);
    __syncthreads();
    h[j] = hn; c[j] = cn;
    hs[(size_t)t * DW + j] = hn;
    __syncthreads();
  }
}

// ---------------- row-wise top-k (LDS-resident row, iterative argmax) -------
__global__ void topk_kernel(const float* __restrict__ rough,
                            float* __restrict__ topf, int* __restrict__ topi) {
  __shared__ float buf[NW];
  __shared__ float rv[256];
  __shared__ int   ri[256];
  const int row = blockIdx.x, tid = threadIdx.x;
  const float* r = rough + (size_t)row * NW;
  for (int jj = tid; jj < NW; jj += 256) buf[jj] = r[jj];
  __syncthreads();
  for (int s = 0; s < KTOP; ++s) {
    float best = -__builtin_inff(); int bi = NW;
    for (int jj = tid; jj < NW; jj += 256) {
      const float v = buf[jj];
      if (v > best || (v == best && jj < bi)) { best = v; bi = jj; }
    }
    rv[tid] = best; ri[tid] = bi;
    __syncthreads();
    for (int st = 128; st > 0; st >>= 1) {
      if (tid < st) {
        const float v2 = rv[tid + st]; const int i2 = ri[tid + st];
        if (v2 > rv[tid] || (v2 == rv[tid] && i2 < ri[tid])) { rv[tid] = v2; ri[tid] = i2; }
      }
      __syncthreads();
    }
    if (tid == 0) {
      int bidx = ri[0]; if (bidx >= NW) bidx = 0;
      topf[(size_t)row * KTOP + s] = rv[0];
      topi[(size_t)row * KTOP + s] = bidx;
      buf[bidx] = -__builtin_inff();
    }
    __syncthreads();
  }
}

// ---------------- fused fine scorer -----------------------------------------
// One block per 32 pair-rows. Features staged in LDS (bf16). hid_W streamed
// as 32-K chunks via TDM double-buffer (tensor_load_to_lds + s_wait_tensorcnt)
// when available; plain global b128 fragment loads otherwise. 8 waves x
// (2 row-tiles x 4 col-tiles) = 32 rows x 512 hidden cols per block, with
// the second FFNN layer fused via shfl reduction.
__global__ void fine_scorer_kernel(const float* __restrict__ words,
                                   const float* __restrict__ topf,
                                   const int* __restrict__ topi,
                                   const float* __restrict__ dist_emb,
                                   const unsigned short* __restrict__ hidwbf,
                                   const float* __restrict__ hid_b,
                                   const float* __restrict__ out_W,
                                   const float* __restrict__ out_b,
                                   float* __restrict__ outf) {
  extern __shared__ __align__(16) char smem_dyn[];
  unsigned short* feats = (unsigned short*)smem_dyn;                    // 32 x FEAT_LD bf16
  unsigned short* bbuf  = (unsigned short*)(smem_dyn + FEATS_BYTES);    // 2 x BCHUNK_ELEMS
  __shared__ float partial[8][TROWS];
  const int tid  = threadIdx.x;
  const int lane = tid & 31, wave = tid >> 5;
  const int tile = blockIdx.x;

  // ---- stage 32 rows x 1600 pair features into LDS as bf16 ----
  {
    const int rl = tid >> 3;      // row in tile, 0..31 (8 threads per row)
    const int cl = tid & 7;
    const int pr = tile * TROWS + rl;
    const int i  = pr / KTOP;
    int b = topi[pr]; if (b < 0 || b >= NW) b = 0;
    int dist = i - b; if (dist < 1) dist = 1;
    int didx;
    if (dist < 5) didx = dist - 1;
    else { int lg = 31 - __clz(dist); if (lg > 6) lg = 6; didx = lg + 2; }
    const float* wa = words + (size_t)i * DW;
    const float* wb = words + (size_t)b * DW;
    const float* de = dist_emb + (size_t)didx * DEMB;
    unsigned short* frow = feats + rl * FEAT_LD;
    for (int cc = cl; cc < PAIRD; cc += 8) {
      float v;
      if (cc < DW)            v = wa[cc];
      else if (cc < 2 * DW)   v = wb[cc - DW];
      else if (cc < 3 * DW)   v = wa[cc - 2 * DW] * wb[cc - 2 * DW];
      else                    v = de[cc - 3 * DW];
      frow[cc] = f2bf(v);
    }
  }
#if HAVE_TDM
  if (wave == 0) tdm_load_hidw_chunk(hidwbf, 0, bbuf);   // prefetch chunk 0
#endif
  __syncthreads();

  // ---- WMMA main loop over K = 1600 in 32-wide chunks ----
  const int colbase = wave * 64;
  v8f acc[2][4] = { { {}, {}, {}, {} }, { {}, {}, {}, {} } };
  const int nchunks = PAIRD / 32;                         // 50
  for (int c = 0; c < nchunks; ++c) {
    const int k0 = c * 32;
#if HAVE_TDM
    if (wave == 0) {
      if (c + 1 < nchunks) {
        tdm_load_hidw_chunk(hidwbf, (c + 1) * 32, bbuf + ((c + 1) & 1) * BCHUNK_ELEMS);
        __builtin_amdgcn_s_wait_tensorcnt(1);   // chunk c landed
      } else {
        __builtin_amdgcn_s_wait_tensorcnt(0);
      }
    }
    __syncthreads();                            // publish chunk c to all waves
    const unsigned short* bsrc = bbuf + (c & 1) * BCHUNK_ELEMS;
    const int bld = 32, bk = 0;
#else
    const unsigned short* bsrc = hidwbf;
    const int bld = PAIRD, bk = k0;
#endif
    const v16bf a0 = load_frag_a(feats, FEAT_LD,  0, k0, lane);
    const v16bf a1 = load_frag_a(feats, FEAT_LD, 16, k0, lane);
#pragma unroll
    for (int t = 0; t < 4; ++t) {
      const v16bf bf = load_frag_b(bsrc, bld, colbase + t * 16, bk, lane);
      acc[0][t] = wmma_bf16(a0, bf, acc[0][t]);
      acc[1][t] = wmma_bf16(a1, bf, acc[1][t]);
    }
#if HAVE_TDM
    __syncthreads();                            // all reads of buf done before reuse
#endif
  }

  // ---- fused epilogue: +bias, LeakyReLU, dot with out_W, lane reduce ----
  const int mbase = (lane >> 4) << 3;
#pragma unroll
  for (int mt = 0; mt < 2; ++mt) {
    float contrib[8] = {0, 0, 0, 0, 0, 0, 0, 0};
#pragma unroll
    for (int t = 0; t < 4; ++t) {
      const int nn = colbase + t * 16 + (lane & 15);
      const float bn = hid_b[nn];
      const float wn = out_W[nn];
#pragma unroll
      for (int r = 0; r < 8; ++r) {
        float hv = acc[mt][t][r] + bn;
        hv = (hv >= 0.f) ? hv : NEG_SLOPE * hv;
        contrib[r] += hv * wn;
      }
    }
#pragma unroll
    for (int r = 0; r < 8; ++r) {
      float v = contrib[r];
      v += __shfl_xor(v, 8);     // reduce across the 16 lanes of each half
      v += __shfl_xor(v, 4);
      v += __shfl_xor(v, 2);
      v += __shfl_xor(v, 1);
      if ((lane & 15) == 0) partial[wave][mt * 16 + mbase + r] = v;
    }
  }
  __syncthreads();

  if (tid < TROWS) {
    float s = out_b[0];
#pragma unroll
    for (int w = 0; w < 8; ++w) s += partial[w][tid];
    const int pr = tile * TROWS + tid;
    const int i = pr / KTOP, kk = pr % KTOP;
    int* outi = (int*)(outf + (size_t)NW * (KTOP + 1));
    outf[(size_t)i * (KTOP + 1) + 1 + kk] = topf[pr] + s;
    if (kk == 0) outf[(size_t)i * (KTOP + 1)] = EPSILONV;
    outi[pr] = topi[pr];
  }
}

// ---------------- host orchestration ---------------------------------------
extern "C" void kernel_launch(void* const* d_in, const int* in_sizes, int n_in,
                              void* d_out, int out_size, void* d_ws, size_t ws_size,
                              hipStream_t stream) {
  const float* word_features = (const float*)d_in[0];
  const float* W_ih  = (const float*)d_in[1];
  const float* W_hh  = (const float*)d_in[2];
  const float* b_ih  = (const float*)d_in[3];
  const float* b_hh  = (const float*)d_in[4];
  const float* bil_W = (const float*)d_in[5];
  const float* bil_b = (const float*)d_in[6];
  const float* dist_emb = (const float*)d_in[7];
  const float* hid_W = (const float*)d_in[8];
  const float* hid_b = (const float*)d_in[9];
  const float* out_W = (const float*)d_in[10];
  const float* out_b = (const float*)d_in[11];
  float* outf = (float*)d_out;

  // workspace carve-up (all chunks 256B-aligned by construction)
  char* w = (char*)d_ws;
  float* xp    = (float*)w;            w += (size_t)NW * G4D * 4;     // 16 MB
  float* hs    = (float*)w;            w += (size_t)NW * DW * 4;      //  4 MB
  float* P     = (float*)w;            w += (size_t)NW * DW * 4;      //  4 MB
  float* rough = (float*)w;            w += (size_t)NW * NW * 4;      // 16 MB
  float* topf  = (float*)w;            w += (size_t)NW * KTOP * 4;
  int*   topi  = (int*)w;              w += (size_t)NW * KTOP * 4;
  float* bsum  = (float*)w;            w += (size_t)G4D * 4;
  unsigned short* xbf    = (unsigned short*)w; w += (size_t)NW * DW * 2;
  unsigned short* wihbf  = (unsigned short*)w; w += (size_t)G4D * DW * 2;
  unsigned short* hsbf   = (unsigned short*)w; w += (size_t)NW * DW * 2;
  unsigned short* pbf    = (unsigned short*)w; w += (size_t)NW * DW * 2;
  unsigned short* bilwbf = (unsigned short*)w; w += (size_t)DW * DW * 2;
  unsigned short* hidwbf = (unsigned short*)w; w += (size_t)HID * PAIRD * 2;

  // bias fold + bf16 pre-conversions (weights converted once, reused by tiles)
  bias_sum_kernel<<<(G4D + 255) / 256, 256, 0, stream>>>(b_ih, b_hh, bsum, G4D);
  f32_to_bf16_kernel<<<(NW * DW + 255) / 256, 256, 0, stream>>>(word_features, xbf, NW * DW);
  f32_to_bf16_kernel<<<(G4D * DW + 255) / 256, 256, 0, stream>>>(W_ih, wihbf, G4D * DW);
  f32_to_bf16_kernel<<<(DW * DW + 255) / 256, 256, 0, stream>>>(bil_W, bilwbf, DW * DW);
  f32_to_bf16_kernel<<<(HID * PAIRD + 255) / 256, 256, 0, stream>>>(hid_W, hidwbf, HID * PAIRD);

  // xp = x @ W_ih^T + (b_ih + b_hh)   [2048 x 2048]
  {
    dim3 g(G4D / 32, NW / 256);
    gemm_nt_bf16_wmma<<<g, 256, 0, stream>>>(xbf, DW, wihbf, DW, bsum, xp, NW, G4D, DW, 0);
  }
  // sequential recurrence -> hs
  lstm_recurrence_kernel<<<1, DW, 0, stream>>>(xp, W_hh, hs);
  f32_to_bf16_kernel<<<(NW * DW + 255) / 256, 256, 0, stream>>>(hs, hsbf, NW * DW);

  // P = words @ bil_W^T + bil_b   [2048 x 512]
  {
    dim3 g(DW / 32, NW / 256);
    gemm_nt_bf16_wmma<<<g, 256, 0, stream>>>(hsbf, DW, bilwbf, DW, bil_b, P, NW, DW, DW, 0);
  }
  f32_to_bf16_kernel<<<(NW * DW + 255) / 256, 256, 0, stream>>>(P, pbf, NW * DW);

  // rough = mask + P @ words^T   [2048 x 2048], strict lower-triangular
  {
    dim3 g(NW / 32, NW / 256);
    gemm_nt_bf16_wmma<<<g, 256, 0, stream>>>(pbf, DW, hsbf, DW, (const float*)nullptr,
                                             rough, NW, NW, DW, 1);
  }

  // top-k per row
  topk_kernel<<<NW, 256, 0, stream>>>(rough, topf, topi);

  // fused fine scorer + output assembly (scores, dummy col, top_idx copy)
  const size_t fine_lds = FEATS_BYTES + 2 * BCHUNK_ELEMS * 2;   // feats + B double-buffer
  fine_scorer_kernel<<<(NW * KTOP) / TROWS, 256, fine_lds, stream>>>(
      hs, topf, topi, dist_emb, hidwbf, hid_b, out_W, out_b, outf);

  (void)in_sizes; (void)n_in; (void)out_size; (void)ws_size;
}
